// AsymG_81260781240672
// MI455X (gfx1250) — matrix operand order repaired and checked by reading, not compiled, not run on hardware
//
#include <hip/hip_runtime.h>

// ---------------------------------------------------------------------------
// MI455X (gfx1250, wave32) implementation.
//
// FLOP budget: edge GEMMs 131 GF (dominant) + self GEMM 13 GF -> bf16 WMMA
// (v_wmma_f32_16x16x32_bf16, fp32 accum).  Finsler edge weights ~33 GF -> fp32
// VALU.  Working set (h: 51MB bf16, emb: 51MB) is L2-resident (192MB), so
// gathers/scatter-atomics hit L2; compute is the bottleneck -> WMMA.
//
// Round-4 change: this toolchain exposes the 6-arg clang-23 form of
// __builtin_amdgcn_tensor_load_to_lds (g0, g1, g2, g3, extra v8i, cpol);
// pass a zero-filled v8i for the extra group.  One TENSOR_LOAD_TO_LDS stages
// the 128 KB packed weight matrix into LDS (2D D#: 1 row x 16384 x 8B),
// issued by wave 0, completed with s_wait_tensorcnt 0 + barrier.
// ---------------------------------------------------------------------------

typedef __attribute__((ext_vector_type(16))) __bf16 v16bf;
typedef __attribute__((ext_vector_type(8)))  float  v8f;
typedef __attribute__((ext_vector_type(4)))  unsigned int tdm_v4u;
typedef __attribute__((ext_vector_type(8)))  int          tdm_v8i;
typedef __attribute__((ext_vector_type(4)))  int          tdm_v4i;

#define N_NODES 100000
#define IN_DIM  256
#define HIDDEN  256
#define EMBED   128
#define N_EDGES 500000
#define BPACK_ELEMS 65536          // 256x256 bf16, WMMA-B packed
#define BPACK_BYTES (BPACK_ELEMS * 2)

static __device__ __forceinline__ unsigned short f2bf_bits(float f) {
  unsigned u = __builtin_bit_cast(unsigned, f);
  u += 0x7FFFu + ((u >> 16) & 1u);            // round-to-nearest-even
  return (unsigned short)(u >> 16);
}

// --------------------------- h (fp32) -> bf16 ------------------------------
__global__ void k_convert_h(const float* __restrict__ h,
                            __bf16* __restrict__ hbf, int n4) {
  int i = blockIdx.x * blockDim.x + threadIdx.x;
  if (i >= n4) return;
  float4 v = ((const float4*)h)[i];
  unsigned long long p =
      (unsigned long long)f2bf_bits(v.x) |
      ((unsigned long long)f2bf_bits(v.y) << 16) |
      ((unsigned long long)f2bf_bits(v.z) << 32) |
      ((unsigned long long)f2bf_bits(v.w) << 48);
  ((unsigned long long*)hbf)[i] = p;
}

// ---- pack W (torch [out,in]) as WMMA B-fragments of B = W^T ([K=256][N=256])
// Layout: Bp[((ntile*8 + kstep)*32 + lane)*16 + i], where
//   lane<16 : N = lane,    K = kstep*32 + (i<8 ? i : 16 + i-8)
//   lane>=16: N = lane-16, K = kstep*32 + 8 + (i<8 ? i : 16 + i-8)
__global__ void k_pack_w(const float* __restrict__ W, __bf16* __restrict__ Bp) {
  int idx  = blockIdx.x * blockDim.x + threadIdx.x;   // 65536 total
  int i    = idx & 15;
  int lane = (idx >> 4) & 31;
  int ks   = (idx >> 9) & 7;
  int nt   = (idx >> 12) & 15;
  int n = nt * 16 + (lane & 15);
  int k = ks * 32 + ((lane >> 4) << 3) + (i & 7) + ((i >> 3) << 4);
  ((unsigned short*)Bp)[idx] = f2bf_bits(W[n * IN_DIM + k]);   // B[k][n]=W[n][k]
}

// ------------------------ Finsler edge weights -----------------------------
// One wave per edge; lane owns dims [4L,4L+3] of 128; x_j broadcast via LDS.
__global__ void __launch_bounds__(256) k_finsler(
    const float* __restrict__ emb, const int* __restrict__ src,
    const int* __restrict__ dst, const float* __restrict__ wbeta,
    const float* __restrict__ Wu, const float* __restrict__ alpha_p,
    float* __restrict__ ew, int E) {
  __shared__ float xs[8][EMBED];
  const int lane = threadIdx.x & 31;
  const int wid  = threadIdx.x >> 5;
  const float alpha = fminf(fmaxf(alpha_p[0], 0.1f), 10.0f);
  int gw = (blockIdx.x * blockDim.x + threadIdx.x) >> 5;
  int nw = (gridDim.x * blockDim.x) >> 5;
  for (int e = gw; e < E; e += nw) {
    const float* xj = emb + (long long)src[e] * EMBED;
    const float* xi = emb + (long long)dst[e] * EMBED;
    float4 a = *(const float4*)(xj + lane * 4);
    float4 b = *(const float4*)(xi + lane * 4);
    *(float4*)&xs[wid][lane * 4] = a;                 // wave-private row
    float d0 = b.x - a.x, d1 = b.y - a.y, d2 = b.z - a.z, d3 = b.w - a.w;
    float ss = d0 * d0 + d1 * d1 + d2 * d2 + d3 * d3;
    float4 wb = *(const float4*)(wbeta + lane * 4);
    float bd = a.x * wb.x + a.y * wb.y + a.z * wb.z + a.w * wb.w;
#pragma unroll
    for (int m = 16; m; m >>= 1) {
      ss += __shfl_xor(ss, m, 32);
      bd += __shfl_xor(bd, m, 32);
    }
    float u0 = 0.f, u1 = 0.f, u2 = 0.f, u3 = 0.f;      // u_j = x_j @ W_u
#pragma unroll 4
    for (int k = 0; k < EMBED; ++k) {
      float xk  = xs[wid][k];                          // LDS broadcast read
      float4 w4 = *(const float4*)(Wu + k * EMBED + lane * 4);
      u0 = fmaf(xk, w4.x, u0); u1 = fmaf(xk, w4.y, u1);
      u2 = fmaf(xk, w4.z, u2); u3 = fmaf(xk, w4.w, u3);
    }
    float as = d0 * u0 + d1 * u1 + d2 * u2 + d3 * u3;
#pragma unroll
    for (int m = 16; m; m >>= 1) as += __shfl_xor(as, m, 32);
    if (lane == 0) {
      float d = sqrtf(ss) + tanhf(bd) * as;
      ew[e] = expf(-alpha * fmaxf(d, 0.0f));
    }
  }
}

// ------------- A-fragment loader (16-bit A 16x32 ISA layout) ---------------
// lane<16: halves 0..7 = K c..c+7, halves 8..15 = K c+16..c+23 with c=0
// lane>=16: same with c=8.  Two b128 loads per fragment.
static __device__ __forceinline__ v16bf load_a_frag(const __bf16* p) {
  v16bf a;
  __builtin_memcpy(&a, p, 16);
  __builtin_memcpy((char*)&a + 16, p + 16, 16);
  return a;
}

// ---- TDM stage of the 128 KB packed-B into LDS (one TENSOR_LOAD_TO_LDS) ---
// D# per CDNA5 ISA: 2D tensor, data_size=8B,
// tensor_dim0 = tile_dim0 = 16384 (x 8B = 128 KB), tile_dim1 = 1.
static __device__ __forceinline__ __bf16* stage_B_tdm(const __bf16* __restrict__ Bp) {
  extern __shared__ __bf16 Bs[];                       // 131072 bytes at launch
  if (threadIdx.x < 32) {                              // wave 0 issues the DMA
    unsigned long long ga = (unsigned long long)Bp;
    unsigned loff = (unsigned)(unsigned long long)(void*)Bs; // LDS byte offset
    tdm_v4u g0;
    g0.x = 1u;                                         // count=1 (valid, user)
    g0.y = loff;                                       // lds_addr
    g0.z = (unsigned)(ga & 0xFFFFFFFFull);             // global_addr[31:0]
    g0.w = (unsigned)((ga >> 32) & 0x1FFFFFFull)       // global_addr[56:32]
           | (2u << 30);                               // type = 2 ("image")
    tdm_v8i g1;
    g1[0] = 3 << 16;          // wg_mask=0, data_size=3 (8 bytes)
    g1[1] = 0x4000 << 16;     // tensor_dim0[15:0]=16384 (in [31:16])
    g1[2] = 1 << 16;          // tensor_dim0[31:16]=0, tensor_dim1[15:0]=1
    g1[3] = 0x4000 << 16;     // tensor_dim1[31:16]=0, tile_dim0=16384
    g1[4] = 1;                // tile_dim1=1, tile_dim2=0
    g1[5] = 16384;            // tensor_dim0_stride[31:0]
    g1[6] = 0;                // stride0[47:32]=0, stride1[15:0]=0
    g1[7] = 0;                // stride1[47:16]=0
    tdm_v4i z4 = {0, 0, 0, 0};                         // groups 2/3 (2D: unused)
    tdm_v8i z8 = {0, 0, 0, 0, 0, 0, 0, 0};             // extra group (clang-23)
    __builtin_amdgcn_tensor_load_to_lds(g0, g1, z4, z4, z8, 0);
    __builtin_amdgcn_s_wait_tensorcnt(0);              // DMA complete
  }
  __syncthreads();                                     // LDS visible to block
  return Bs;
}

// ------------------ edge aggregation: gather-GEMM-scatter ------------------
// One wave per 16-edge tile.  D[16x256] = hbf[src] @ B, then row-scale by
// w[e] after adding bias, atomic scatter-add into out[dst[e]].
__global__ void __launch_bounds__(256) k_edge_gemm(
    const __bf16* __restrict__ hbf, const __bf16* __restrict__ Bp,
    const float* __restrict__ bias, const int* __restrict__ src,
    const int* __restrict__ dst, const float* __restrict__ wgt,
    float* __restrict__ out, int ntiles) {
  const __bf16* Bs = stage_B_tdm(Bp);
  const int lane = threadIdx.x & 31;
  const int half = lane >> 4;
  const int l15  = lane & 15;
  const int cOff = half * 8;
  int gw = (blockIdx.x * blockDim.x + threadIdx.x) >> 5;
  int nw = (gridDim.x * blockDim.x) >> 5;
  for (int t = gw; t < ntiles; t += nw) {
    const long long srow = src[t * 16 + l15];          // A-matrix row M=l15
    const __bf16* arow = hbf + srow * IN_DIM;
    v16bf A[8];
#pragma unroll
    for (int ks = 0; ks < 8; ++ks) A[ks] = load_a_frag(arow + ks * 32 + cOff);
    // C/D layout: VGPR r -> M = half*8 + r, N = l15
    float wv[8]; long long drow[8];
    const int mBase = t * 16 + half * 8;
#pragma unroll
    for (int r = 0; r < 8; ++r) {
      wv[r]   = wgt[mBase + r];
      drow[r] = (long long)dst[mBase + r] * HIDDEN;
    }
#pragma unroll 2
    for (int nt = 0; nt < 16; ++nt) {
      v8f acc = {};
      const __bf16* bp = Bs + ((size_t)(nt * 8) * 32 + lane) * 16;
#pragma unroll
      for (int ks = 0; ks < 8; ++ks) {
        v16bf B;
        __builtin_memcpy(&B, bp + (size_t)ks * 512, 32);   // 2x ds_load_b128
        acc = __builtin_amdgcn_wmma_f32_16x16x32_bf16(
            false, A[ks], false, B, (short)0, acc, false, false);
      }
      const int n = nt * 16 + l15;
      const float bn = bias[n];
#pragma unroll
      for (int r = 0; r < 8; ++r)
        atomicAdd(out + drow[r] + n, wv[r] * (acc[r] + bn));
    }
  }
}

// --------------------- self GEMM: out = h @ Wself^T + b --------------------
__global__ void __launch_bounds__(256) k_self_gemm(
    const __bf16* __restrict__ hbf, const __bf16* __restrict__ Bp,
    const float* __restrict__ bias, float* __restrict__ out, int ntiles) {
  const __bf16* Bs = stage_B_tdm(Bp);
  const int lane = threadIdx.x & 31;
  const int half = lane >> 4;
  const int l15  = lane & 15;
  const int cOff = half * 8;
  int gw = (blockIdx.x * blockDim.x + threadIdx.x) >> 5;
  int nw = (gridDim.x * blockDim.x) >> 5;
  for (int t = gw; t < ntiles; t += nw) {
    const __bf16* arow = hbf + (size_t)(t * 16 + l15) * IN_DIM;
    v16bf A[8];
#pragma unroll
    for (int ks = 0; ks < 8; ++ks) A[ks] = load_a_frag(arow + ks * 32 + cOff);
    const size_t mBase = (size_t)t * 16 + half * 8;
#pragma unroll 2
    for (int nt = 0; nt < 16; ++nt) {
      v8f acc = {};
      const __bf16* bp = Bs + ((size_t)(nt * 8) * 32 + lane) * 16;
#pragma unroll
      for (int ks = 0; ks < 8; ++ks) {
        v16bf B;
        __builtin_memcpy(&B, bp + (size_t)ks * 512, 32);   // 2x ds_load_b128
        acc = __builtin_amdgcn_wmma_f32_16x16x32_bf16(
            false, A[ks], false, B, (short)0, acc, false, false);
      }
      const int n = nt * 16 + l15;
      const float bn = bias[n];
#pragma unroll
      for (int r = 0; r < 8; ++r)
        out[(mBase + r) * HIDDEN + n] = acc[r] + bn;   // initializes out
    }
  }
}

// -------------------------------- ReLU -------------------------------------
__global__ void k_relu(float* __restrict__ out, int n4) {
  int i = blockIdx.x * blockDim.x + threadIdx.x;
  if (i >= n4) return;
  float4 v = ((float4*)out)[i];
  v.x = fmaxf(v.x, 0.f); v.y = fmaxf(v.y, 0.f);
  v.z = fmaxf(v.z, 0.f); v.w = fmaxf(v.w, 0.f);
  ((float4*)out)[i] = v;
}

// ---------------------------------------------------------------------------
extern "C" void kernel_launch(void* const* d_in, const int* in_sizes, int n_in,
                              void* d_out, int out_size, void* d_ws,
                              size_t ws_size, hipStream_t stream) {
  const float* h          = (const float*)d_in[0];
  const int*   pos_src    = (const int*)d_in[1];
  const int*   pos_dst    = (const int*)d_in[2];
  const int*   neg_src    = (const int*)d_in[3];
  const int*   neg_dst    = (const int*)d_in[4];
  const float* emb        = (const float*)d_in[5];
  const float* W_pos      = (const float*)d_in[6];
  const float* b_pos      = (const float*)d_in[7];
  const float* W_neg      = (const float*)d_in[8];
  const float* b_neg      = (const float*)d_in[9];
  const float* W_self     = (const float*)d_in[10];
  const float* b_self     = (const float*)d_in[11];
  const float* w_pos_beta = (const float*)d_in[12];
  const float* W_pos_u    = (const float*)d_in[13];
  const float* alpha_pos  = (const float*)d_in[14];
  const float* w_neg_beta = (const float*)d_in[15];
  const float* W_neg_u    = (const float*)d_in[16];
  const float* alpha_neg  = (const float*)d_in[17];
  float* out = (float*)d_out;

  // workspace carve (~55.6 MB total)
  char* ws = (char*)d_ws;
  size_t off = 0;
  auto carve = [&](size_t bytes) -> char* {
    char* p = ws + off;
    off += (bytes + 255) & ~(size_t)255;
    return p;
  };
  __bf16* hbf     = (__bf16*)carve((size_t)N_NODES * IN_DIM * 2);  // 51.2 MB
  __bf16* Bp_pos  = (__bf16*)carve(BPACK_BYTES);
  __bf16* Bp_neg  = (__bf16*)carve(BPACK_BYTES);
  __bf16* Bp_self = (__bf16*)carve(BPACK_BYTES);
  float*  pw      = (float*)carve((size_t)N_EDGES * 4);
  float*  nw      = (float*)carve((size_t)N_EDGES * 4);

  const int n4h = N_NODES * IN_DIM / 4;
  k_convert_h<<<(n4h + 255) / 256, 256, 0, stream>>>(h, hbf, n4h);
  k_pack_w<<<256, 256, 0, stream>>>(W_pos, Bp_pos);
  k_pack_w<<<256, 256, 0, stream>>>(W_neg, Bp_neg);
  k_pack_w<<<256, 256, 0, stream>>>(W_self, Bp_self);

  k_finsler<<<2048, 256, 0, stream>>>(emb, pos_src, pos_dst, w_pos_beta,
                                      W_pos_u, alpha_pos, pw, N_EDGES);
  k_finsler<<<2048, 256, 0, stream>>>(emb, neg_src, neg_dst, w_neg_beta,
                                      W_neg_u, alpha_neg, nw, N_EDGES);

  const int selfTiles = N_NODES / 16;                   // 6250 exact
  k_self_gemm<<<(selfTiles + 15) / 16, 256, BPACK_BYTES, stream>>>(
      hbf, Bp_self, b_self, out, selfTiles);            // ~2 tiles / wave

  const int edgeTiles = N_EDGES / 16;                   // 31250 exact
  k_edge_gemm<<<(edgeTiles + 15) / 16, 256, BPACK_BYTES, stream>>>(
      hbf, Bp_pos, b_pos, pos_src, pos_dst, pw, out, edgeTiles);
  k_edge_gemm<<<(edgeTiles + 15) / 16, 256, BPACK_BYTES, stream>>>(
      hbf, Bp_neg, b_neg, neg_src, neg_dst, nw, out, edgeTiles);

  const int n4o = N_NODES * HIDDEN / 4;
  k_relu<<<(n4o + 255) / 256, 256, 0, stream>>>(out, n4o);
}
